// Attention_73486890434886
// MI455X (gfx1250) — matrix operand order repaired and checked by reading, not compiled, not run on hardware
//
#include <hip/hip_runtime.h>
#include <hip/hip_bf16.h>

// ---------------------------------------------------------------------------
// Fused gated multi-head attention for MI455X (gfx1250, wave32, WMMA).
// Shapes: B=1, Q_LEN=KV_LEN=2048, Q_DIM=KV_DIM=256, N_HEAD=8, C=32, OUT=256.
// Memory-bound on the 134MB bias stream; all other tensors are L2-resident.
// All GEMMs use v_wmma_f32_16x16x32_f16 with contiguous f16 fragment loads;
// K/V tiles arrive in LDS via global_load_async_to_lds_b128 (ASYNCcnt).
// ---------------------------------------------------------------------------

typedef __attribute__((ext_vector_type(16))) _Float16 v16h;
typedef __attribute__((ext_vector_type(8)))  _Float16 v8h;
typedef __attribute__((ext_vector_type(8)))  float    v8f;

#define QLEN 2048
#define NHEAD 8
#define CDIM 32

__device__ __forceinline__ v8f wmma16(v16h a, v16h b, v8f c) {
  // D = A(16x32 f16) * B(32x16 f16) + C(16x16 f32)
  return __builtin_amdgcn_wmma_f32_16x16x32_f16(
      false, a, false, b, (short)0, c, false, false);
}

__device__ __forceinline__ v16h cat8(v8h lo, v8h hi) {
  v16h r;
#pragma unroll
  for (int i = 0; i < 8; ++i) { r[i] = lo[i]; r[8 + i] = hi[i]; }
  return r;
}

// Load an A/B fragment whose 16 half elements are two contiguous 16B chunks.
__device__ __forceinline__ v16h frag2x8(const _Float16* p0, const _Float16* p1) {
  return cat8(*(const v8h*)p0, *(const v8h*)p1);
}

// ===========================================================================
// Kernel 0a: convert q/kv inputs f32 -> f16 (read once, 4MB total).
// grid = (2048, 2); block = 256 (one row per block, one element per thread).
// ===========================================================================
__global__ __launch_bounds__(256) void prep_x(
    const float* __restrict__ Xq, const float* __restrict__ Xkv,
    _Float16* __restrict__ Xq_h, _Float16* __restrict__ Xkv_h) {
  const size_t i = (size_t)blockIdx.x * 256 + threadIdx.x;
  if (blockIdx.y == 0) Xq_h[i]  = (_Float16)Xq[i];
  else                 Xkv_h[i] = (_Float16)Xkv[i];
}

// ===========================================================================
// Kernel 0b: one-shot weight prep. Convert f32 weights to f16 and transpose
// to a B-fragment-friendly [col][contraction] layout (contraction contiguous).
// grid = (512, 3); block = 256 (one contraction slice per thread).
// ===========================================================================
__global__ __launch_bounds__(256) void prep_w(
    const float* __restrict__ qgw, const float* __restrict__ kvw,
    const float* __restrict__ ow,
    _Float16* __restrict__ wq_t, _Float16* __restrict__ wkv_t,
    _Float16* __restrict__ wo_t) {
  const int col = blockIdx.x, z = blockIdx.y, a = threadIdx.x;
  if (z == 0)      wq_t [(size_t)col * 256 + a] = (_Float16)qgw[(size_t)a * 512 + col];
  else if (z == 1) wkv_t[(size_t)col * 256 + a] = (_Float16)kvw[(size_t)a * 512 + col];
  else if (col < 256)
                   wo_t [(size_t)col * 256 + a] = (_Float16)ow [(size_t)a * 256 + col];
}

// ===========================================================================
// Kernel 1: fused q/g and k/v projections (all-f16 fragments).
//   Y[row, h*64+c] = sum_a X[row,a] * W[a, h*64+c] + Wb[h*64+c]
// grid = (16 row-blocks, 8 heads, 2 sources); block = 256 (8 waves).
// Outputs: q (f16, pre-scaled), k (f16), sigmoid(g) (f32),
//          v TRANSPOSED (f16, [h][c][row]) for contiguous PV B-fragments.
// ===========================================================================
__global__ __launch_bounds__(256) void proj_kernel(
    const _Float16* __restrict__ Xq_h, const _Float16* __restrict__ Xkv_h,
    const _Float16* __restrict__ WqT, const _Float16* __restrict__ WkvT,
    const float* __restrict__ Bq, const float* __restrict__ Bkv,
    _Float16* __restrict__ q_h, _Float16* __restrict__ k_h,
    _Float16* __restrict__ v_t, float* __restrict__ g_s) {
  const int blk = blockIdx.x, h = blockIdx.y;
  const int isQ = (blockIdx.z == 0);
  const _Float16* __restrict__ X  = isQ ? Xq_h : Xkv_h;
  const _Float16* __restrict__ WT = isQ ? WqT : WkvT;
  const float*    __restrict__ Wb = isQ ? Bq : Bkv;

  const int tid = threadIdx.x, wave = tid >> 5, lane = tid & 31;
  const int ln = lane & 15, hg = lane >> 4;
  const int rowbase = blk * 128 + wave * 16;
  const _Float16* xrow = X + (size_t)(rowbase + ln) * 256;

  v8f accs[4];
#pragma unroll
  for (int nf = 0; nf < 4; ++nf) {
    const float bv = Wb[h * 64 + nf * 16 + ln];
#pragma unroll
    for (int v = 0; v < 8; ++v) accs[nf][v] = bv;
  }

  for (int kb = 0; kb < 8; ++kb) {
    const int abase = kb * 32;
    // A fragment: row m = ln; K chunk [abase+hg*16,+16)
    const v16h a = frag2x8(xrow + abase + hg * 16, xrow + abase + hg * 16 + 8);
#pragma unroll
    for (int nf = 0; nf < 4; ++nf) {
      const int col = h * 64 + nf * 16 + ln;
      const _Float16* bp = WT + (size_t)col * 256 + abase + hg * 16;
      accs[nf] = wmma16(a, frag2x8(bp, bp + 8), accs[nf]);
    }
  }

  const float qscale = 0.17677669529663687f;  // 1/sqrt(C)
#pragma unroll
  for (int nf = 0; nf < 4; ++nf) {
    const int cc = nf * 16 + ln;  // 0..63 within head
#pragma unroll
    for (int v = 0; v < 8; ++v) {
      const int row = rowbase + v + 8 * hg;
      const float y = accs[nf][v];
      if (cc < 32) {
        if (isQ) q_h[((size_t)h * QLEN + row) * CDIM + cc] = (_Float16)(y * qscale);
        else     k_h[((size_t)h * QLEN + row) * CDIM + cc] = (_Float16)y;
      } else {
        const int c2 = cc - 32;
        if (isQ) g_s[((size_t)h * QLEN + row) * CDIM + c2] = 1.0f / (1.0f + __expf(-y));
        else     v_t[((size_t)h * CDIM + c2) * QLEN + row] = (_Float16)y;  // transposed
      }
    }
  }
}

// ===========================================================================
// Kernel 2: flash attention (streams the 134MB bias exactly once).
// grid = (16 q-blocks, 8 heads); block = 256 (8 waves). Wave owns 16 q rows.
// K tile (128x32) and transposed V tile (32x128) land in LDS via
// global_load_async_to_lds_b128; P round-trips via wave-private LDS.
// ===========================================================================
#define KTS 40    // K-tile padded row stride (halfs): 128 rows
#define VTS 136   // V^T-tile padded row stride (halfs): 32 rows
#define PTS 136   // P-buffer padded row stride (halfs)
#define VT_OFF (128 * KTS)            // halfs
#define PB_OFF (VT_OFF + 32 * VTS)    // halfs
#define SMEM_HALFS (PB_OFF + 8 * 16 * PTS)

__device__ __forceinline__ void async_lds_32B(unsigned lds_byte_off,
                                              const _Float16* gptr) {
  const unsigned long long ga = (unsigned long long)(const void*)gptr;
  asm volatile("global_load_async_to_lds_b128 %0, %1, off"
               :: "v"(lds_byte_off), "v"(ga) : "memory");
  asm volatile("global_load_async_to_lds_b128 %0, %1, off offset:16"
               :: "v"(lds_byte_off), "v"(ga) : "memory");
}

__global__ __launch_bounds__(256) void attn_kernel(
    const _Float16* __restrict__ q_h, const _Float16* __restrict__ k_h,
    const _Float16* __restrict__ v_t, const float* __restrict__ g_s,
    const float* __restrict__ bias, _Float16* __restrict__ ao_h) {
  __shared__ __attribute__((aligned(32))) _Float16 smem[SMEM_HALFS];

  const int blk = blockIdx.x, h = blockIdx.y;
  const int tid = threadIdx.x, wave = tid >> 5, lane = tid & 31;
  const int ln = lane & 15, hg = lane >> 4;
  const int rowbase = blk * 128 + wave * 16;

  // Q A-fragment (pre-scaled by 1/sqrt(C)): m = ln, K = c (32).
  v16h aq;
  {
    const _Float16* qrow = q_h + ((size_t)h * QLEN + rowbase + ln) * CDIM;
    aq = frag2x8(qrow + hg * 16, qrow + hg * 16 + 8);
  }

  v8f acc0 = {}, acc1 = {};
  float mi[8], li[8];
#pragma unroll
  for (int v = 0; v < 8; ++v) { mi[v] = -__builtin_inff(); li[v] = 0.0f; }

  _Float16* myP = smem + PB_OFF + wave * 16 * PTS;
  const float* bias_h = bias + (size_t)h * QLEN * QLEN;

  // Per-thread async-copy coordinates (constant across chunks).
  const int kr = tid >> 1, kpart = tid & 1;          // K tile: 128 rows x 64B
  const int vc = tid >> 3, vseg = tid & 7;           // V^T tile: 32 rows x 128 halfs
  const unsigned kldsoff = (unsigned)((kr * KTS + kpart * 16) * 2);
  const unsigned vldsoff = (unsigned)((VT_OFF + vc * VTS + vseg * 16) * 2);

  for (int kc = 0; kc < QLEN; kc += 128) {
    // --- async global -> LDS tile fill (tracked by ASYNCcnt) ---
    async_lds_32B(kldsoff, k_h + ((size_t)h * QLEN + kc + kr) * CDIM + kpart * 16);
    async_lds_32B(vldsoff, v_t + ((size_t)h * CDIM + vc) * QLEN + kc + vseg * 16);
    asm volatile("s_wait_asynccnt 0x0" ::: "memory");
    __syncthreads();

    // --- S = Q K^T + bias : 8 fragments of 16x16 (bias loaded into C) ---
    v8f s[8];
#pragma unroll
    for (int nb = 0; nb < 8; ++nb) {
      v8f c;
      const float* bp = bias_h + (size_t)rowbase * QLEN + kc + nb * 16 + ln;
#pragma unroll
      for (int v = 0; v < 8; ++v) c[v] = bp[(size_t)(v + 8 * hg) * QLEN];
      const _Float16* kp = smem + (nb * 16 + ln) * KTS + hg * 16;
      s[nb] = wmma16(aq, frag2x8(kp, kp + 8), c);
    }

    // --- online softmax (row reductions across 16 lanes of the half-wave)
    float rm[8];
#pragma unroll
    for (int v = 0; v < 8; ++v) {
      float r = s[0][v];
#pragma unroll
      for (int nb = 1; nb < 8; ++nb) r = fmaxf(r, s[nb][v]);
      r = fmaxf(r, __shfl_xor(r, 1, 32));
      r = fmaxf(r, __shfl_xor(r, 2, 32));
      r = fmaxf(r, __shfl_xor(r, 4, 32));
      r = fmaxf(r, __shfl_xor(r, 8, 32));
      rm[v] = r;
    }
    float alpha[8];
#pragma unroll
    for (int v = 0; v < 8; ++v) {
      const float mn = fmaxf(mi[v], rm[v]);
      alpha[v] = __expf(mi[v] - mn);
      mi[v] = mn;
    }
    float rs[8];
#pragma unroll
    for (int v = 0; v < 8; ++v) rs[v] = 0.0f;
#pragma unroll
    for (int nb = 0; nb < 8; ++nb) {
#pragma unroll
      for (int v = 0; v < 8; ++v) {
        const float p = __expf(s[nb][v] - mi[v]);
        rs[v] += p;
        myP[(v + 8 * hg) * PTS + nb * 16 + ln] = (_Float16)p;
      }
    }
#pragma unroll
    for (int v = 0; v < 8; ++v) {
      float r = rs[v];
      r += __shfl_xor(r, 1, 32);
      r += __shfl_xor(r, 2, 32);
      r += __shfl_xor(r, 4, 32);
      r += __shfl_xor(r, 8, 32);
      li[v] = li[v] * alpha[v] + r;
      acc0[v] *= alpha[v];
      acc1[v] *= alpha[v];
    }

    // --- O += P V : contiguous A (from P buffer) and B (from V^T tile) ---
#pragma unroll
    for (int ks = 0; ks < 4; ++ks) {
      const _Float16* pp = myP + ln * PTS + ks * 32;
      v16h ap = frag2x8(pp + hg * 16, pp + hg * 16 + 8);
#pragma unroll
      for (int nbv = 0; nbv < 2; ++nbv) {
        // b[i] -> kv = ks*32 + hg*16 + i (contiguous in V^T row), c = nbv*16+ln
        const _Float16* vp = smem + VT_OFF + (nbv * 16 + ln) * VTS + ks * 32 + hg * 16;
        if (nbv == 0) acc0 = wmma16(ap, frag2x8(vp, vp + 8), acc0);
        else          acc1 = wmma16(ap, frag2x8(vp, vp + 8), acc1);
      }
    }
    __syncthreads();
  }

  // epilogue: normalize, gate with sigmoid(g), store f16 (q, h*32+c) layout
#pragma unroll
  for (int v = 0; v < 8; ++v) {
    const int row = rowbase + v + 8 * hg;
    const float inv = 1.0f / li[v];
    const float* gp = g_s + ((size_t)h * QLEN + row) * CDIM;
    ao_h[(size_t)row * 256 + h * 32 + ln]      = (_Float16)(acc0[v] * inv * gp[ln]);
    ao_h[(size_t)row * 256 + h * 32 + 16 + ln] = (_Float16)(acc1[v] * inv * gp[16 + ln]);
  }
}

// ===========================================================================
// Kernel 3: output GEMM  out[q,o] = ao(2048x256 f16) @ o_w + o_b[o]
// grid = (16 row-blocks, 4 col-blocks); wave: 16 rows x 64 cols.
// B fragments come from the pre-transposed f16 wo_t: 2 contiguous loads each.
// ===========================================================================
__global__ __launch_bounds__(256) void out_kernel(
    const _Float16* __restrict__ ao_h, const _Float16* __restrict__ wo_t,
    const float* __restrict__ o_b, float* __restrict__ out) {
  const int blk = blockIdx.x, nblk = blockIdx.y;
  const int tid = threadIdx.x, wave = tid >> 5, lane = tid & 31;
  const int ln = lane & 15, hg = lane >> 4;
  const int rowbase = blk * 128 + wave * 16;

  v8f accs[4];
#pragma unroll
  for (int nf = 0; nf < 4; ++nf) {
    const float bv = o_b[nblk * 64 + nf * 16 + ln];
#pragma unroll
    for (int v = 0; v < 8; ++v) accs[nf][v] = bv;
  }

  for (int kb = 0; kb < 8; ++kb) {
    const _Float16* arow = ao_h + (size_t)(rowbase + ln) * 256 + kb * 32;
    v16h a = frag2x8(arow + hg * 16, arow + hg * 16 + 8);
#pragma unroll
    for (int nf = 0; nf < 4; ++nf) {
      const int col = nblk * 64 + nf * 16 + ln;
      const _Float16* bp = wo_t + (size_t)col * 256 + kb * 32 + hg * 16;
      accs[nf] = wmma16(a, frag2x8(bp, bp + 8), accs[nf]);
    }
  }

#pragma unroll
  for (int nf = 0; nf < 4; ++nf) {
    const int col = nblk * 64 + nf * 16 + ln;
#pragma unroll
    for (int v = 0; v < 8; ++v)
      out[(size_t)(rowbase + v + 8 * hg) * 256 + col] = accs[nf][v];
  }
}

// ===========================================================================
extern "C" void kernel_launch(void* const* d_in, const int* in_sizes, int n_in,
                              void* d_out, int out_size, void* d_ws, size_t ws_size,
                              hipStream_t stream) {
  const float* q_inputs   = (const float*)d_in[0];  // (1,2048,256)
  const float* kv_inputs  = (const float*)d_in[1];  // (1,2048,256)
  const float* bias       = (const float*)d_in[2];  // (1,8,2048,2048)
  const float* qg_weights = (const float*)d_in[3];  // (256,1,8,64)
  const float* qg_bias    = (const float*)d_in[4];  // (1,8,1,64)
  const float* kv_weights = (const float*)d_in[5];  // (256,1,8,64)
  const float* kv_bias    = (const float*)d_in[6];  // (1,8,1,64)
  const float* o_weights  = (const float*)d_in[7];  // (1,8,32,256)
  const float* o_bias     = (const float*)d_in[8];  // (256,1)
  float* out = (float*)d_out;                       // (2048,256)

  // Workspace carve-out (~9MB)
  const size_t HQC = (size_t)NHEAD * QLEN * CDIM;   // 512K elements
  const size_t XSZ = (size_t)QLEN * 256;            // 512K elements
  _Float16* q_h    = (_Float16*)d_ws;
  _Float16* k_h    = q_h + HQC;
  _Float16* v_t    = k_h + HQC;                  // transposed: [h][c][row]
  _Float16* ao_h   = v_t + HQC;
  _Float16* xq_h   = ao_h + XSZ;                 // f16 copy of q_inputs
  _Float16* xkv_h  = xq_h + XSZ;                 // f16 copy of kv_inputs
  _Float16* wq_t   = xkv_h + XSZ;                // [512][256]
  _Float16* wkv_t  = wq_t + 512 * 256;           // [512][256]
  _Float16* wo_t   = wkv_t + 512 * 256;          // [256][256]
  float*    g_s    = (float*)(wo_t + 256 * 256);

  prep_x<<<dim3(2048, 2), 256, 0, stream>>>(q_inputs, kv_inputs, xq_h, xkv_h);

  prep_w<<<dim3(512, 3), 256, 0, stream>>>(qg_weights, kv_weights, o_weights,
                                           wq_t, wkv_t, wo_t);

  proj_kernel<<<dim3(16, 8, 2), 256, 0, stream>>>(
      xq_h, xkv_h, wq_t, wkv_t, qg_bias, kv_bias, q_h, k_h, v_t, g_s);

  attn_kernel<<<dim3(16, 8), 256, 0, stream>>>(q_h, k_h, v_t, g_s, bias, ao_h);

  out_kernel<<<dim3(16, 4), 256, 0, stream>>>(ao_h, wo_t, o_bias, out);
}
